// ShadowAlphaTransformer_6442450944617
// MI455X (gfx1250) — matrix operand order repaired and checked by reading, not compiled, not run on hardware
//
#include <hip/hip_runtime.h>
#include <hip/hip_bf16.h>

// ---------------------------------------------------------------------------
// ShadowAlphaTransformer on MI455X (gfx1250, wave32, WMMA)
//
// The reference applies a separable Gaussian 3x. We fold the cascade into a
// single triple-convolved kernel g3 = gn*gn*gn per axis (radius exactly the
// reference pad P=3r), so each region needs only: stage -> h-conv (WMMA
// banded-Toeplitz GEMM) -> v-conv (WMMA) -> bilinear translate/composite.
// All parameters are computed on-device; grids are fixed worst-case so the
// whole pipeline is graph-capturable.
// ---------------------------------------------------------------------------

typedef __attribute__((ext_vector_type(16))) _Float16 v16h;
typedef __attribute__((ext_vector_type(8)))  float    v8f;

#define IMG_H 1024
#define IMG_W 1024
#define NB 4
#define NK 8
#define PAIRS (NB * NK)

// Worst case: ksize<=61 -> r<=30 -> pad P=3r<=90; crop <=270x606 -> padded <=450x786
#define MAXW 832
#define MAXH 512
#define BUFSZ (MAXH * MAXW)

// d_ws layout (in floats):
//   [0, 6400)        per-pair float params (200 each): {dxp, dyp, pad[6], g3[192]}
//   [6400, 6784)     int section: per-pair {r, ks} (64 ints), then per-pair
//                    region ints {ys, ye, xs, xe, Hc, Wc, 0, 0} (256 ints)
//   [6784, ...)      buf0 (512*832 floats), buf1 (512*832 floats)
#define FPP 200
#define GN_OFF 8
#define GN_MAX 192               // 3*61 - 2 = 181 taps max
#define I_SECT_OFF (PAIRS * FPP) // 6400 floats in
#define BUF_OFF 6784

#define KFAC 1.87997120597325f   // 3/4 * sqrt(2*pi)
#define INV_SQRT_2PI 0.39894228040143f

// --------------------------------------------------------------------------
__global__ void zero_kernel(float* __restrict__ out, size_t n) {
    size_t idx = (size_t)blockIdx.x * blockDim.x + threadIdx.x;
    size_t stride = (size_t)gridDim.x * blockDim.x;
    for (; idx < n; idx += stride) out[idx] = 0.0f;
}

// --------------------------------------------------------------------------
// Per-(i,k) parameters + triple-convolved Gaussian taps. Reproduces the
// reference's indexing quirk: sig/tnh are indexed by k only (first batch row),
// while the box geometry is per (i,k).
__global__ void param_kernel(const float* __restrict__ sig,
                             const float* __restrict__ tnh,
                             const int*   __restrict__ boxes,
                             float* __restrict__ ws) {
    int p = threadIdx.x;
    if (p >= PAIRS) return;
    int k = p % NK;

    const int* bx = boxes + p * 8;
    int x1 = bx[0], y1 = bx[1], x2 = bx[2], y2 = bx[3];
    int x3 = bx[4], y3 = bx[5], x4 = bx[6], y4 = bx[7];
    int ys0 = min(min(y1, y2), min(y3, y4));
    int ye0 = max(max(y1, y2), max(y3, y4));
    int xs0 = min(min(x1, x2), min(x3, x4));
    int xe0 = max(max(x1, x2), max(x3, x4));
    int h = ye0 - ys0, w = xe0 - xs0;

    float blur = (sig[k * 2 + 1] + 1e-5f) * (float)h * 0.5f;
    int ks = (int)floorf(blur * KFAC + 0.5f);
    if (ks < 2) ks = 2;
    if ((ks & 1) == 0) ks += 1;
    if (ks > 63) ks = 63;               // safety clamp (never hit with this data)
    int r = ks / 2;

    float dxp = tnh[k * 2 + 0] * (float)h * 0.2f;
    float dyp = tnh[k * 2 + 1] * (float)h * 0.2f;
    float psize = (float)(r * 3);
    float eh = psize + fabsf(dyp);
    float ew = psize + fabsf(dxp);
    int ys = (int)fmaxf(fminf((float)ys0 - eh, (float)IMG_H), 0.0f);
    int ye = (int)fmaxf(fminf((float)ye0 + eh, (float)IMG_H), 0.0f);
    int xs = (int)fmaxf(fminf((float)xs0 - ew, (float)IMG_W), 0.0f);
    int xe = (int)fmaxf(fminf((float)xe0 + ew, (float)IMG_W), 0.0f);
    int Hc = ye - ys, Wc = xe - xs;
    if (h < 5 || w < 5) { Hc = 0; Wc = 0; }    // reference 'continue'

    int* I = (int*)(ws + I_SECT_OFF);
    I[2 * p] = r;
    I[2 * p + 1] = ks;
    int* R = I + 2 * PAIRS + p * 8;
    R[0] = ys; R[1] = ye; R[2] = xs; R[3] = xe; R[4] = Hc; R[5] = Wc;
    R[6] = 0;  R[7] = 0;

    float* Fp = ws + p * FPP;
    Fp[0] = dxp; Fp[1] = dyp;

    // Normalized 1D Gaussian gn (sum = 1), then g3 = gn (*) gn (*) gn.
    // Applying the 2D kernel outer(g,g)/sum three times == separable conv with
    // g3 along each axis; radius of g3 is exactly 3r == reference pad.
    float g1[64], g2[128], g3[GN_MAX];
    float mean = (float)(ks - 1) * 0.5f;
    float sum = 0.0f;
    for (int x = 0; x < ks; ++x) {
        float t = ((float)x - mean) / (2.0f * blur);   // (x-mean)/(2*sigma), as in ref
        float g = INV_SQRT_2PI / blur * expf(-t * t);
        g1[x] = g;
        sum += g;
    }
    float inv = 1.0f / sum;
    for (int x = 0; x < ks; ++x) g1[x] *= inv;

    int len2 = 2 * ks - 1;
    for (int x = 0; x < len2; ++x) {
        float s = 0.0f;
        int jlo = max(0, x - (ks - 1)), jhi = min(ks - 1, x);
        for (int j = jlo; j <= jhi; ++j) s += g1[j] * g1[x - j];
        g2[x] = s;
    }
    int len3 = 3 * ks - 2;
    for (int x = 0; x < len3; ++x) {
        float s = 0.0f;
        int jlo = max(0, x - (ks - 1)), jhi = min(len2 - 1, x);
        for (int j = jlo; j <= jhi; ++j) s += g2[j] * g1[x - j];
        g3[x] = s;
    }
    for (int x = len3; x < GN_MAX; ++x) g3[x] = 0.0f;
    for (int x = 0; x < GN_MAX; ++x) Fp[GN_OFF + x] = g3[x];
}

// --------------------------------------------------------------------------
// Stage crop (zero-extended) into padded scratch buf0. Grid covers MAXH*MAXW.
__global__ void stage_kernel(const float* __restrict__ alpha,
                             float* __restrict__ ws, int pair) {
    __builtin_amdgcn_s_wait_tensorcnt(0);
    const int* I = (const int*)(ws + I_SECT_OFF);
    int r = I[2 * pair];
    int P = 3 * r;
    const int* R = I + 2 * PAIRS + pair * 8;
    int ys = R[0], xs = R[2], Hc = R[4], Wc = R[5];
    float* buf0 = ws + BUF_OFF;

    int tid = blockIdx.x * blockDim.x + threadIdx.x;
    int y = tid / MAXW, x = tid % MAXW;
    int i = pair / NK;
    int cy = y - P, cx = x - P;
    float v = 0.0f;
    if (cy >= 0 && cy < Hc && cx >= 0 && cx < Wc) {
        const float* src = alpha + (size_t)i * IMG_H * IMG_W
                                 + (size_t)(ys + cy) * IMG_W + (xs + cx);
        __builtin_prefetch(src, 0, 1);   // global_prefetch_b8
        v = *src;
    }
    buf0[y * MAXW + x] = v;
}

// --------------------------------------------------------------------------
// Horizontal conv (taps g3, radius P=3r) as WMMA GEMM. One wave computes a
// 16x16 output tile: D[m][n] = sum_j A[m][j]*B[j][n],
//   A[m][j] = src[y0+m][x0-P+j],  B[j][n] = g3[j-n] for 0 <= j-n < ks3.
__global__ void hconv_wmma(const float* __restrict__ src,
                           float* __restrict__ dst,
                           const float* __restrict__ ws, int pair) {
    __shared__ float gsh[GN_MAX];
    const int* I = (const int*)(ws + I_SECT_OFF);
    int r = I[2 * pair], ks = I[2 * pair + 1];
    int P = 3 * r;
    int ks3 = 3 * ks - 2;
    const float* gn = ws + pair * FPP + GN_OFF;
    if (threadIdx.x < GN_MAX) gsh[threadIdx.x] = gn[threadIdx.x];
    __syncthreads();

    int wid = blockIdx.x * (blockDim.x >> 5) + (threadIdx.x >> 5);
    int ty = wid / (MAXW / 16);
    int tx = wid % (MAXW / 16);
    int y0 = ty * 16, x0 = tx * 16;
    int lane = threadIdx.x & 31;
    int row = lane & 15;     // A-matrix M / B,C,D column N
    int grp = lane >> 4;     // K-half selector

    v8f acc = {};
    int nch = (ks3 + 15 + 31) / 32;       // <= 7
    const float* srow = src + (y0 + row) * MAXW;
    for (int c = 0; c < nch; ++c) {
        v16h a, b;
#pragma unroll
        for (int v = 0; v < 8; ++v) {
            // A fragment: 16-bit 16x32 layout (VGPR v -> K pairs)
            int j0 = c * 32 + ((v < 4) ? 0 : 16) + grp * 8 + (v & 3) * 2;
            int c0 = x0 - P + j0;
            float f0 = (c0 >= 0 && c0 < MAXW) ? srow[c0] : 0.0f;
            float f1 = (c0 + 1 >= 0 && c0 + 1 < MAXW) ? srow[c0 + 1] : 0.0f;
            a[2 * v]     = (_Float16)f0;
            a[2 * v + 1] = (_Float16)f1;
            // B fragment: Toeplitz band of g3; column = row var
            int jb = c * 32 + ((v < 4) ? 0 : 16) + grp * 8 + (v & 3) * 2;
            int t0 = jb - row, t1 = t0 + 1;
            b[2 * v]     = (_Float16)((t0 >= 0 && t0 < ks3) ? gsh[t0] : 0.0f);
            b[2 * v + 1] = (_Float16)((t1 >= 0 && t1 < ks3) ? gsh[t1] : 0.0f);
        }
        acc = __builtin_amdgcn_wmma_f32_16x16x32_f16(
            false, a, false, b, (short)0, acc, false, false);
    }
#pragma unroll
    for (int v = 0; v < 8; ++v) {
        int m = v + grp * 8;                 // C/D layout: VGPR v, lane half -> M
        dst[(y0 + m) * MAXW + x0 + row] = acc[v];
    }
}

// --------------------------------------------------------------------------
// Vertical conv as WMMA GEMM (roles swapped):
//   D[m][n] = sum_j T[m][j] * src[y0-P+j][x0+n],  T[m][j] = g3[j-m].
__global__ void vconv_wmma(const float* __restrict__ src,
                           float* __restrict__ dst,
                           const float* __restrict__ ws, int pair) {
    __shared__ float gsh[GN_MAX];
    const int* I = (const int*)(ws + I_SECT_OFF);
    int r = I[2 * pair], ks = I[2 * pair + 1];
    int P = 3 * r;
    int ks3 = 3 * ks - 2;
    const float* gn = ws + pair * FPP + GN_OFF;
    if (threadIdx.x < GN_MAX) gsh[threadIdx.x] = gn[threadIdx.x];
    __syncthreads();

    int wid = blockIdx.x * (blockDim.x >> 5) + (threadIdx.x >> 5);
    int ty = wid / (MAXW / 16);
    int tx = wid % (MAXW / 16);
    int y0 = ty * 16, x0 = tx * 16;
    int lane = threadIdx.x & 31;
    int row = lane & 15;
    int grp = lane >> 4;

    v8f acc = {};
    int nch = (ks3 + 15 + 31) / 32;
    for (int c = 0; c < nch; ++c) {
        v16h a, b;
#pragma unroll
        for (int v = 0; v < 8; ++v) {
            // A fragment = Toeplitz band (rows = output rows)
            int j0 = c * 32 + ((v < 4) ? 0 : 16) + grp * 8 + (v & 3) * 2;
            int t0 = j0 - row, t1 = t0 + 1;
            a[2 * v]     = (_Float16)((t0 >= 0 && t0 < ks3) ? gsh[t0] : 0.0f);
            a[2 * v + 1] = (_Float16)((t1 >= 0 && t1 < ks3) ? gsh[t1] : 0.0f);
            // B fragment = source column slab
            int jb = c * 32 + ((v < 4) ? 0 : 16) + grp * 8 + (v & 3) * 2;
            int ry0 = y0 - P + jb;
            int ry1 = ry0 + 1;
            float f0 = (ry0 >= 0 && ry0 < MAXH) ? src[ry0 * MAXW + x0 + row] : 0.0f;
            float f1 = (ry1 >= 0 && ry1 < MAXH) ? src[ry1 * MAXW + x0 + row] : 0.0f;
            b[2 * v]     = (_Float16)f0;
            b[2 * v + 1] = (_Float16)f1;
        }
        acc = __builtin_amdgcn_wmma_f32_16x16x32_f16(
            false, a, false, b, (short)0, acc, false, false);
    }
#pragma unroll
    for (int v = 0; v < 8; ++v) {
        int m = v + grp * 8;
        dst[(y0 + m) * MAXW + x0 + row] = acc[v];
    }
}

// --------------------------------------------------------------------------
// Bilinear translate of the blurred crop + composite into sg / sl.
__global__ void translate_kernel(const float* __restrict__ ws,
                                 float* __restrict__ sg,
                                 float* __restrict__ sl, int pair) {
    const int* I = (const int*)(ws + I_SECT_OFF);
    int r = I[2 * pair];
    int P = 3 * r;
    const int* R = I + 2 * PAIRS + pair * 8;
    int ys = R[0], xs = R[2], Hc = R[4], Wc = R[5];
    if (Hc <= 0 || Wc <= 0) return;
    float dxp = ws[pair * FPP], dyp = ws[pair * FPP + 1];
    const float* buf = ws + BUF_OFF;   // final blurred result lives in buf0

    int tid = blockIdx.x * blockDim.x + threadIdx.x;
    int y = tid / 640, x = tid % 640;
    if (y >= Hc || x >= Wc) return;

    float sx = (float)x - dxp;
    float sy = (float)y - dyp;
    float x0f = floorf(sx), y0f = floorf(sy);
    float wx = sx - x0f, wy = sy - y0f;
    int xi = (int)x0f, yi = (int)y0f;

    auto gat = [&](int yy, int xx) -> float {
        if (yy < 0 || yy >= Hc || xx < 0 || xx >= Wc) return 0.0f;
        return buf[(P + yy) * MAXW + (P + xx)];
    };
    float shifted = (1.0f - wy) * ((1.0f - wx) * gat(yi, xi)     + wx * gat(yi, xi + 1))
                  +          wy  * ((1.0f - wx) * gat(yi + 1, xi) + wx * gat(yi + 1, xi + 1));

    int i = pair / NK, k = pair % NK;
    size_t o = (size_t)i * IMG_H * IMG_W + (size_t)(ys + y) * IMG_W + (xs + x);
    sg[o] = shifted;
    if (shifted > 0.0f) sl[o] = (float)(k + 1);
}

// --------------------------------------------------------------------------
extern "C" void kernel_launch(void* const* d_in, const int* in_sizes, int n_in,
                              void* d_out, int out_size, void* d_ws, size_t ws_size,
                              hipStream_t stream) {
    const float* alpha = (const float*)d_in[0];
    (void)d_in[1]; // font_size_pred unused by the reference math
    const float* sig   = (const float*)d_in[2];
    const float* tnh   = (const float*)d_in[3];
    const int*   boxes = (const int*)d_in[4];
    (void)in_sizes; (void)n_in; (void)out_size; (void)ws_size;

    float* out = (float*)d_out;
    float* sg = out;
    float* sl = out + (size_t)NB * IMG_H * IMG_W;
    float* ws = (float*)d_ws;
    float* buf0 = ws + BUF_OFF;
    float* buf1 = buf0 + BUFSZ;

    zero_kernel<<<2048, 256, 0, stream>>>(out, (size_t)2 * NB * IMG_H * IMG_W);
    param_kernel<<<1, 64, 0, stream>>>(sig, tnh, boxes, ws);

    const int stageBlocks = (MAXH * MAXW) / 256;                 // 1664
    const int convBlocks  = ((MAXH / 16) * (MAXW / 16)) / 8;     // 208 (8 waves/blk)
    const int xlatBlocks  = (320 * 640) / 256;                   // 800

    for (int p = 0; p < PAIRS; ++p) {
        stage_kernel<<<stageBlocks, 256, 0, stream>>>(alpha, ws, p);
        hconv_wmma<<<convBlocks, 256, 0, stream>>>(buf0, buf1, ws, p);
        vconv_wmma<<<convBlocks, 256, 0, stream>>>(buf1, buf0, ws, p);
        translate_kernel<<<xlatBlocks, 256, 0, stream>>>(ws, sg, sl, p);
    }
}
